// CausalSelfAttention_27032524161702
// MI455X (gfx1250) — matrix operand order repaired and checked by reading, not compiled, not run on hardware
//
#include <hip/hip_runtime.h>
#include <hip/hip_bf16.h>

typedef __attribute__((ext_vector_type(16))) __bf16 bf16x16;
typedef __attribute__((ext_vector_type(8)))  __bf16 bf16x8;
typedef __attribute__((ext_vector_type(8)))  float  f32x8;
typedef __attribute__((ext_vector_type(4)))  unsigned int u32x4;
typedef __attribute__((ext_vector_type(8)))  int   i32x8;
typedef __attribute__((ext_vector_type(4)))  int   i32x4;

#define N_EMBED 1024
#define N_HEAD  16
#define HS      64
#define BATCH   4
#define SEQ     2048
#define MROWS   (BATCH * SEQ)
#define KCHUNKS 32   // K = 1024 in 32-wide chunks

static __device__ __forceinline__ f32x8 wmma_bf16(bf16x16 a, bf16x16 b, f32x8 c) {
  return __builtin_amdgcn_wmma_f32_16x16x32_bf16(false, a, false, b, (short)0, c,
                                                 false, false);
}

// ---------------------------------------------------------------------------
// Tensor Data Mover: 1-D contiguous copy global -> LDS (ndwords 4-byte units).
// ---------------------------------------------------------------------------
static __device__ __forceinline__ void tdm_load_1d(unsigned int lds_off,
                                                   const void* gsrc,
                                                   int ndwords) {
  unsigned long long ga = (unsigned long long)(uintptr_t)gsrc;
  u32x4 g0;
  g0[0] = 1u;                                    // count=1, user descriptor
  g0[1] = lds_off;                               // lds_addr
  g0[2] = (unsigned int)ga;                      // global_addr[31:0]
  g0[3] = (unsigned int)(ga >> 32) | (2u << 30); // global_addr[56:32] | type=2
  i32x8 g1;
  g1[0] = (2 << 16);                             // data_size = 4B
  g1[1] = (int)((ndwords & 0xFFFF) << 16);       // tensor_dim0 lo16
  g1[2] = (int)(((ndwords >> 16) & 0xFFFF) | (1 << 16)); // dim0 hi | tensor_dim1=1
  g1[3] = (int)(((unsigned int)ndwords) << 16);  // tile_dim0 = ndwords
  g1[4] = 1;                                     // tile_dim1 = 1
  g1[5] = ndwords;                               // tensor_dim0_stride
  g1[6] = 0;
  g1[7] = 0;
  i32x4 gz = {0, 0, 0, 0};
#if __clang_major__ >= 23
  i32x8 gz8 = {0, 0, 0, 0, 0, 0, 0, 0};
  __builtin_amdgcn_tensor_load_to_lds(g0, g1, gz, gz, gz8, 0);
#else
  __builtin_amdgcn_tensor_load_to_lds(g0, g1, gz, gz, 0);
#endif
}

// ---------------------------------------------------------------------------
// Fragment-major swizzled layouts ([..tile..][lane][16 contiguous bf16]).
// A-frag (16x32): lane = m%16 + 16*((k>>4)&1), j = k&15
// B-frag (32x16): lane = n%16 + 16*((k>>4)&1), j = k&15
// ---------------------------------------------------------------------------
static __device__ __forceinline__ size_t a_swz_idx(int m, int k) {
  int lane = (m & 15) + (((k >> 4) & 1) << 4);
  int j = k & 15;
  return ((((size_t)(m >> 6) * KCHUNKS + (k >> 5)) * 4 + ((m >> 4) & 3)) << 9)
         + lane * 16 + j;
}
// weights: [ntile][kchunk][panel] so tiles are contiguous across k for fixed ntile
static __device__ __forceinline__ size_t b_swz_idx(int k, int n) {
  int lane = (n & 15) + (((k >> 4) & 1) << 4);
  int j = k & 15;
  return ((((size_t)(n >> 7) * KCHUNKS + (k >> 5)) * 8 + ((n >> 4) & 7)) << 9)
         + lane * 16 + j;
}
static __device__ __forceinline__ size_t q_swz_idx(int bh, int t, int d) {
  int lane = (t & 15) + (((d >> 4) & 1) << 4);
  int j = d & 15;
  return ((((size_t)bh * 128 + (t >> 4)) * 2 + (d >> 5)) << 9) + lane * 16 + j;
}
static __device__ __forceinline__ size_t k_swz_idx(int bh, int t, int d) {
  int lane = (t & 15) + (((d >> 4) & 1) << 4);
  int j = d & 15;
  return ((((size_t)bh * 128 + (t >> 4)) * 2 + (d >> 5)) << 9) + lane * 16 + j;
}
static __device__ __forceinline__ size_t v_swz_idx(int bh, int t, int d) {
  int lane = (d & 15) + (((t >> 4) & 1) << 4);
  int j = t & 15;
  return ((((size_t)bh * 64 + (t >> 5)) * 4 + (d >> 4)) << 9) + lane * 16 + j;
}

// ---------------------------------------------------------------------------
// Converters: f32 row-major -> bf16 fragment-swizzled. 8 consecutive k for a
// fixed row land in 8 contiguous swizzled slots -> one 16B store per thread.
// ---------------------------------------------------------------------------
__global__ __launch_bounds__(256) void cvt_a_swz(const float* __restrict__ src,
                                                 __bf16* __restrict__ dst) {
  int idx = blockIdx.x * 256 + threadIdx.x;   // MROWS*128 threads
  int m = idx >> 7;
  int kc8 = (idx & 127) << 3;
  bf16x8 v;
#pragma unroll
  for (int j = 0; j < 8; ++j) v[j] = (__bf16)src[(size_t)m * N_EMBED + kc8 + j];
  *(bf16x8*)&dst[a_swz_idx(m, kc8)] = v;
}

__global__ __launch_bounds__(256) void cvt_b_swz(const float* __restrict__ src,
                                                 __bf16* __restrict__ dst, int N) {
  int n = blockIdx.x * 256 + threadIdx.x;     // grid (N/256, 128)
  int kc8 = blockIdx.y << 3;
  bf16x8 v;
#pragma unroll
  for (int j = 0; j < 8; ++j) v[j] = (__bf16)src[(size_t)(kc8 + j) * N + n];
  *(bf16x8*)&dst[b_swz_idx(kc8, n)] = v;
}

// ---------------------------------------------------------------------------
// Kernel 1: qkv = x @ w_attn + b_attn -> q/k/v fragment-swizzled.
// WG tile 64x128, 8 waves x (32x32). K-step 64, double-buffered TDM staging.
// ---------------------------------------------------------------------------
__global__ __launch_bounds__(256) void qkv_gemm(const __bf16* __restrict__ xa,
                                                const __bf16* __restrict__ wa,
                                                const float* __restrict__ bias,
                                                __bf16* __restrict__ qb,
                                                __bf16* __restrict__ kb,
                                                __bf16* __restrict__ vb) {
  __shared__ __align__(32) __bf16 As[2 * 4096];   // 2 x (64x64 A tile)
  __shared__ __align__(32) __bf16 Bs[2 * 8192];   // 2 x (64x128 B tile)

  int tid  = threadIdx.x;
  int lane = tid & 31;
  int wave = tid >> 5;
  int mtile = blockIdx.x;
  int ntile = blockIdx.y;
  int sub0 = (wave & 1) * 2;
  int pan0 = (wave >> 1) * 2;
  unsigned int as_off = (unsigned int)(uintptr_t)(void*)As;
  unsigned int bs_off = (unsigned int)(uintptr_t)(void*)Bs;
  const bf16x16* afr = (const bf16x16*)As;
  const bf16x16* bfr = (const bf16x16*)Bs;

  const __bf16* asrc = xa + ((size_t)mtile * KCHUNKS) * 2048;  // 2048 elems / chunk
  const __bf16* bsrc = wa + ((size_t)ntile * KCHUNKS) * 4096;  // 4096 elems / chunk

  f32x8 acc00 = {}, acc01 = {}, acc10 = {}, acc11 = {};

  if (wave == 0)      tdm_load_1d(as_off, asrc, 2048);
  else if (wave == 1) tdm_load_1d(bs_off, bsrc, 4096);
  __builtin_amdgcn_s_wait_tensorcnt(0);
  __syncthreads();

  for (int s = 0; s < 16; ++s) {         // 16 steps x 64 K
    int cur = s & 1, nxt = cur ^ 1;
    if (s + 1 < 16) {
      if (wave == 0)
        tdm_load_1d(as_off + nxt * 8192, asrc + (size_t)(s + 1) * 4096, 2048);
      else if (wave == 1)
        tdm_load_1d(bs_off + nxt * 16384, bsrc + (size_t)(s + 1) * 8192, 4096);
    }
    int ab = cur * 256, bb = cur * 512;  // bf16x16 units
#pragma unroll
    for (int k2 = 0; k2 < 2; ++k2) {
      bf16x16 a0 = afr[ab + (k2 * 4 + sub0) * 32 + lane];
      bf16x16 a1 = afr[ab + (k2 * 4 + sub0 + 1) * 32 + lane];
      bf16x16 b0 = bfr[bb + (k2 * 8 + pan0) * 32 + lane];
      bf16x16 b1 = bfr[bb + (k2 * 8 + pan0 + 1) * 32 + lane];
      acc00 = wmma_bf16(a0, b0, acc00);
      acc01 = wmma_bf16(a0, b1, acc01);
      acc10 = wmma_bf16(a1, b0, acc10);
      acc11 = wmma_bf16(a1, b1, acc11);
    }
    __builtin_amdgcn_s_wait_tensorcnt(0);
    __syncthreads();
  }

  int half = lane >> 4, nn = lane & 15;
#pragma unroll
  for (int ti = 0; ti < 2; ++ti) {
#pragma unroll
    for (int tj = 0; tj < 2; ++tj) {
      f32x8 a = (ti == 0) ? (tj == 0 ? acc00 : acc01) : (tj == 0 ? acc10 : acc11);
#pragma unroll
      for (int r = 0; r < 8; ++r) {
        int row = mtile * 64 + (wave & 1) * 32 + ti * 16 + half * 8 + r;
        int col = ntile * 128 + (wave >> 1) * 32 + tj * 16 + nn;
        float val = a[r] + bias[col];
        int which = col >> 10;              // 0=q 1=k 2=v
        int c = col & (N_EMBED - 1);
        int h = c >> 6, d = c & 63;
        int b = row >> 11, t = row & (SEQ - 1);
        int bh = b * N_HEAD + h;
        if (which == 0)       qb[q_swz_idx(bh, t, d)] = (__bf16)val;
        else if (which == 1)  kb[k_swz_idx(bh, t, d)] = (__bf16)val;
        else                  vb[v_swz_idx(bh, t, d)] = (__bf16)val;
      }
    }
  }
}

// ---------------------------------------------------------------------------
// Kernel 2: fused causal flash attention. WG = 128 query rows (8 waves x 16).
// 64-key K/V blocks, double-buffered TDM staging.
// ---------------------------------------------------------------------------
__global__ __launch_bounds__(256) void attn_fused(const __bf16* __restrict__ qg,
                                                  const __bf16* __restrict__ kg,
                                                  const __bf16* __restrict__ vg,
                                                  __bf16* __restrict__ yg) {
  __shared__ __align__(32) __bf16 Kt[2 * 4096];
  __shared__ __align__(32) __bf16 Vt[2 * 4096];
  __shared__ __align__(32) __bf16 Ps[8 * 1024];   // per-wave P, A-frag layout

  int tid = threadIdx.x, lane = tid & 31, wave = tid >> 5;
  int half = lane >> 4, l4 = lane & 15;
  int bh = blockIdx.y;
  int qbase = blockIdx.x * 128;
  int qr = qbase + wave * 16;
  unsigned int kt_off = (unsigned int)(uintptr_t)(void*)Kt;
  unsigned int vt_off = (unsigned int)(uintptr_t)(void*)Vt;
  const __bf16* ksrc = kg + ((size_t)bh * 128) * 1024;  // 4096 elems / 64 keys
  const __bf16* vsrc = vg + ((size_t)bh * 64) * 2048;

  // Q fragments: direct global b128 loads from swizzled layout
  const bf16x16* qfr =
      (const bf16x16*)(qg + ((((size_t)bh * 128 + (qr >> 4)) * 2) << 9));
  bf16x16 aq0 = qfr[lane];
  bf16x16 aq1 = qfr[32 + lane];

  f32x8 o0 = {}, o1 = {}, o2 = {}, o3 = {};
  float rm[8], rl[8];
#pragma unroll
  for (int r = 0; r < 8; ++r) { rm[r] = -1e30f; rl[r] = 0.0f; }
  __bf16* ps = &Ps[wave * 1024];
  const bf16x16* pfr = (const bf16x16*)ps;

  int nblocks = (qbase + 128) >> 6;
  if (wave == 0)      tdm_load_1d(kt_off, ksrc, 2048);
  else if (wave == 1) tdm_load_1d(vt_off, vsrc, 2048);
  __builtin_amdgcn_s_wait_tensorcnt(0);
  __syncthreads();

  for (int bk = 0; bk < nblocks; ++bk) {
    int cur = bk & 1, nxt = cur ^ 1;
    if (bk + 1 < nblocks) {
      if (wave == 0)
        tdm_load_1d(kt_off + nxt * 8192, ksrc + (size_t)(bk + 1) * 4096, 2048);
      else if (wave == 1)
        tdm_load_1d(vt_off + nxt * 8192, vsrc + (size_t)(bk + 1) * 4096, 2048);
    }
    int kb = bk * 64;
    const bf16x16* kfr = (const bf16x16*)Kt + cur * 256;
    const bf16x16* vfr = (const bf16x16*)Vt + cur * 256;

    // S = Q K^T * 1/sqrt(hs), causal mask
    f32x8 s[4];
#pragma unroll
    for (int t4 = 0; t4 < 4; ++t4) {
      bf16x16 b0 = kfr[(t4 * 2 + 0) * 32 + lane];
      bf16x16 b1 = kfr[(t4 * 2 + 1) * 32 + lane];
      f32x8 z = {};
      z = wmma_bf16(aq0, b0, z);
      z = wmma_bf16(aq1, b1, z);
#pragma unroll
      for (int r = 0; r < 8; ++r) {
        int rowq = qr + half * 8 + r;
        int colk = kb + t4 * 16 + l4;
        float sv = z[r] * 0.125f;
        z[r] = (colk <= rowq) ? sv : -1e30f;
      }
      s[t4] = z;
    }

    // online softmax
    float alpha[8];
#pragma unroll
    for (int r = 0; r < 8; ++r) {
      float mx = fmaxf(fmaxf(s[0][r], s[1][r]), fmaxf(s[2][r], s[3][r]));
      mx = fmaxf(mx, __shfl_xor(mx, 1));
      mx = fmaxf(mx, __shfl_xor(mx, 2));
      mx = fmaxf(mx, __shfl_xor(mx, 4));
      mx = fmaxf(mx, __shfl_xor(mx, 8));
      float nm = fmaxf(rm[r], mx);
      alpha[r] = __expf(rm[r] - nm);
      rm[r] = nm;
    }
#pragma unroll
    for (int t4 = 0; t4 < 4; ++t4)
#pragma unroll
      for (int r = 0; r < 8; ++r) s[t4][r] = __expf(s[t4][r] - rm[r]);
#pragma unroll
    for (int r = 0; r < 8; ++r) {
      float sum = (s[0][r] + s[1][r]) + (s[2][r] + s[3][r]);
      sum += __shfl_xor(sum, 1);
      sum += __shfl_xor(sum, 2);
      sum += __shfl_xor(sum, 4);
      sum += __shfl_xor(sum, 8);
      rl[r] = rl[r] * alpha[r] + sum;
      o0[r] *= alpha[r]; o1[r] *= alpha[r]; o2[r] *= alpha[r]; o3[r] *= alpha[r];
    }

    // P -> per-wave LDS scratch in A-frag layout
#pragma unroll
    for (int t4 = 0; t4 < 4; ++t4) {
      int chunk = t4 >> 1;
      int jj = l4;
      int laneA16 = (t4 & 1) << 4;
#pragma unroll
      for (int r = 0; r < 8; ++r) {
        int mrow = half * 8 + r;
        ps[chunk * 512 + (mrow + laneA16) * 16 + jj] = (__bf16)s[t4][r];
      }
    }
    bf16x16 ap0 = pfr[lane];
    bf16x16 ap1 = pfr[32 + lane];

    // O += P @ V
    o0 = wmma_bf16(ap0, vfr[(0 * 4 + 0) * 32 + lane], o0);
    o0 = wmma_bf16(ap1, vfr[(1 * 4 + 0) * 32 + lane], o0);
    o1 = wmma_bf16(ap0, vfr[(0 * 4 + 1) * 32 + lane], o1);
    o1 = wmma_bf16(ap1, vfr[(1 * 4 + 1) * 32 + lane], o1);
    o2 = wmma_bf16(ap0, vfr[(0 * 4 + 2) * 32 + lane], o2);
    o2 = wmma_bf16(ap1, vfr[(1 * 4 + 2) * 32 + lane], o2);
    o3 = wmma_bf16(ap0, vfr[(0 * 4 + 3) * 32 + lane], o3);
    o3 = wmma_bf16(ap1, vfr[(1 * 4 + 3) * 32 + lane], o3);

    __builtin_amdgcn_s_wait_tensorcnt(0);
    __syncthreads();
  }

  // normalize; store y in A-frag swizzled layout (feeds proj GEMM)
  int b = bh >> 4, h = bh & 15;
#pragma unroll
  for (int r = 0; r < 8; ++r) {
    float inv = 1.0f / rl[r];
    int m = b * SEQ + qr + half * 8 + r;
#pragma unroll
    for (int dt = 0; dt < 4; ++dt) {
      int c = h * HS + dt * 16 + l4;
      float v = (dt == 0 ? o0[r] : dt == 1 ? o1[r] : dt == 2 ? o2[r] : o3[r]);
      yg[a_swz_idx(m, c)] = (__bf16)(v * inv);
    }
  }
}

// ---------------------------------------------------------------------------
// Kernel 3: out = y @ w_proj + b_proj (f32). Same double-buffered structure.
// ---------------------------------------------------------------------------
__global__ __launch_bounds__(256) void proj_gemm(const __bf16* __restrict__ ya,
                                                 const __bf16* __restrict__ wp,
                                                 const float* __restrict__ bias,
                                                 float* __restrict__ out) {
  __shared__ __align__(32) __bf16 As[2 * 4096];
  __shared__ __align__(32) __bf16 Bs[2 * 8192];

  int tid  = threadIdx.x;
  int lane = tid & 31;
  int wave = tid >> 5;
  int mtile = blockIdx.x;
  int ntile = blockIdx.y;
  int sub0 = (wave & 1) * 2;
  int pan0 = (wave >> 1) * 2;
  unsigned int as_off = (unsigned int)(uintptr_t)(void*)As;
  unsigned int bs_off = (unsigned int)(uintptr_t)(void*)Bs;
  const bf16x16* afr = (const bf16x16*)As;
  const bf16x16* bfr = (const bf16x16*)Bs;

  const __bf16* asrc = ya + ((size_t)mtile * KCHUNKS) * 2048;
  const __bf16* bsrc = wp + ((size_t)ntile * KCHUNKS) * 4096;

  f32x8 acc00 = {}, acc01 = {}, acc10 = {}, acc11 = {};

  if (wave == 0)      tdm_load_1d(as_off, asrc, 2048);
  else if (wave == 1) tdm_load_1d(bs_off, bsrc, 4096);
  __builtin_amdgcn_s_wait_tensorcnt(0);
  __syncthreads();

  for (int s = 0; s < 16; ++s) {
    int cur = s & 1, nxt = cur ^ 1;
    if (s + 1 < 16) {
      if (wave == 0)
        tdm_load_1d(as_off + nxt * 8192, asrc + (size_t)(s + 1) * 4096, 2048);
      else if (wave == 1)
        tdm_load_1d(bs_off + nxt * 16384, bsrc + (size_t)(s + 1) * 8192, 4096);
    }
    int ab = cur * 256, bb = cur * 512;
#pragma unroll
    for (int k2 = 0; k2 < 2; ++k2) {
      bf16x16 a0 = afr[ab + (k2 * 4 + sub0) * 32 + lane];
      bf16x16 a1 = afr[ab + (k2 * 4 + sub0 + 1) * 32 + lane];
      bf16x16 b0 = bfr[bb + (k2 * 8 + pan0) * 32 + lane];
      bf16x16 b1 = bfr[bb + (k2 * 8 + pan0 + 1) * 32 + lane];
      acc00 = wmma_bf16(a0, b0, acc00);
      acc01 = wmma_bf16(a0, b1, acc01);
      acc10 = wmma_bf16(a1, b0, acc10);
      acc11 = wmma_bf16(a1, b1, acc11);
    }
    __builtin_amdgcn_s_wait_tensorcnt(0);
    __syncthreads();
  }

  int half = lane >> 4, nn = lane & 15;
#pragma unroll
  for (int ti = 0; ti < 2; ++ti) {
#pragma unroll
    for (int tj = 0; tj < 2; ++tj) {
      f32x8 a = (ti == 0) ? (tj == 0 ? acc00 : acc01) : (tj == 0 ? acc10 : acc11);
#pragma unroll
      for (int r = 0; r < 8; ++r) {
        int row = mtile * 64 + (wave & 1) * 32 + ti * 16 + half * 8 + r;
        int col = ntile * 128 + (wave >> 1) * 32 + tj * 16 + nn;
        out[(size_t)row * N_EMBED + col] = a[r] + bias[col];
      }
    }
  }
}

// ---------------------------------------------------------------------------
extern "C" void kernel_launch(void* const* d_in, const int* in_sizes, int n_in,
                              void* d_out, int out_size, void* d_ws, size_t ws_size,
                              hipStream_t stream) {
  const float* x      = (const float*)d_in[0];
  const float* w_attn = (const float*)d_in[1];
  const float* b_attn = (const float*)d_in[2];
  const float* w_proj = (const float*)d_in[3];
  const float* b_proj = (const float*)d_in[4];
  float* out = (float*)d_out;

  size_t nq = (size_t)BATCH * N_HEAD * SEQ * HS;          // 8,388,608
  __bf16* xa = (__bf16*)d_ws;                             // 16 MB
  __bf16* wa = xa + (size_t)MROWS * N_EMBED;              // 6 MB
  __bf16* wp = wa + (size_t)N_EMBED * 3 * N_EMBED;        // 2 MB
  __bf16* qb = wp + (size_t)N_EMBED * N_EMBED;            // 16 MB
  __bf16* kb = qb + nq;                                   // 16 MB
  __bf16* vb = kb + nq;                                   // 16 MB
  __bf16* yb = vb + nq;                                   // 16 MB (total 88 MB)

  cvt_a_swz<<<(MROWS * 128) / 256, 256, 0, stream>>>(x, xa);
  cvt_b_swz<<<dim3((3 * N_EMBED) / 256, 128), 256, 0, stream>>>(w_attn, wa,
                                                                3 * N_EMBED);
  cvt_b_swz<<<dim3(N_EMBED / 256, 128), 256, 0, stream>>>(w_proj, wp, N_EMBED);

  qkv_gemm<<<dim3(MROWS / 64, (3 * N_EMBED) / 128), 256, 0, stream>>>(
      xa, wa, b_attn, qb, kb, vb);
  attn_fused<<<dim3(SEQ / 128, BATCH * N_HEAD), 256, 0, stream>>>(qb, kb, vb, yb);
  proj_gemm<<<dim3(MROWS / 64, N_EMBED / 128), 256, 0, stream>>>(
      yb, wp, b_proj, out);
}
